// Attention_87471303950744
// MI455X (gfx1250) — matrix operand order repaired and checked by reading, not compiled, not run on hardware
//
#include <hip/hip_runtime.h>
#include <hip/hip_bf16.h>

// ---------------------------------------------------------------------------
// CDNA5 (gfx1250) multi-head attention forward, bf16 WMMA / f32 accumulate,
// async Global->LDS tile staging (inline asm, ASYNCcnt) where the data path
// allows a pure copy.  B=2, S=2048, D=1024, H=16, dh=64.
// ---------------------------------------------------------------------------

typedef __attribute__((ext_vector_type(16))) __bf16 v16bf;
typedef __attribute__((ext_vector_type(8)))  float  v8f;

#define SEQ    2048
#define DMODEL 1024
#define NHEAD  16
#define HDIM   64
#define BATCH  2
#define MROWS  (BATCH * SEQ)   // 4096

// ---------------- async Global->LDS copy (gfx1250 ASYNCcnt path) ------------
// GLOBAL_LOAD_ASYNC_TO_LDS_B128 (ISA 15.18.3, op 98), GV addressing:
//   vdst  = per-lane LDS byte address (flat addr truncated to 32 bits),
//   vaddr = 64-bit global address, saddr = off.
// Tracked by ASYNCcnt; completed with s_wait_asynccnt + barrier.
__device__ __forceinline__ void copy16_to_lds(void* ldsp, const void* gp) {
  const unsigned lds_off = (unsigned)(unsigned long long)ldsp;
  const unsigned long long ga = (unsigned long long)gp;
  asm volatile("global_load_async_to_lds_b128 %0, %1, off"
               :
               : "v"(lds_off), "v"(ga)
               : "memory");
}

__device__ __forceinline__ void wait_async_copies() {
  asm volatile("s_wait_asynccnt 0" ::: "memory");
}

// -------------------------- WMMA fragment helpers --------------------------
// A fragment: 16(M) x 32(K) bf16 from row-major LDS tile (ldt elems/row, even).
// Lane maps M = lane&15; dword j holds K pair base = half*16 + 2j (contiguous k order, as in B).
__device__ __forceinline__ v16bf load_frag_a(const __bf16* tile, int ldt,
                                             int row0, int col0) {
  const int lane = threadIdx.x & 31;
  const int half = lane >> 4, r = lane & 15;
  const __bf16* rp = tile + (size_t)(row0 + r) * ldt + col0 + half * 16;
  union { v16bf v; unsigned u[8]; } f;
#pragma unroll
  for (int j = 0; j < 8; ++j) {
    const int kb = 2 * j;
    f.u[j] = *(const unsigned*)(rp + kb);            // two adjacent bf16
  }
  return f.v;
}

// B fragment: 32(K) x 16(N) bf16 read from a TRANSPOSED tile Bt[N][K]
// (K-pair per dword contiguous in LDS).
// VGPR j, lanes0-15 -> K=2j,2j+1 ; lanes16-31 -> K=16+2j,16+2j+1 (contiguous, as in A).
__device__ __forceinline__ v16bf load_frag_b(const __bf16* tileT, int ldt,
                                             int n0, int k0) {
  const int lane = threadIdx.x & 31;
  const int half = lane >> 4, r = lane & 15;
  const __bf16* rp = tileT + (size_t)(n0 + r) * ldt + k0 + half * 16;
  union { v16bf v; unsigned u[8]; } f;
#pragma unroll
  for (int j = 0; j < 8; ++j) f.u[j] = *(const unsigned*)(rp + 2 * j);
  return f.v;
}

__device__ __forceinline__ v8f wmma_bf16(v16bf a, v16bf b, v8f c) {
  // 8 args: (neg_a, A, neg_b, B, c_mod, C, reuse_a, reuse_b)
  return __builtin_amdgcn_wmma_f32_16x16x32_bf16(false, a, false, b,
                                                 (short)0, c, false, false);
}

__device__ __forceinline__ float red_max16(float v) {
  v = fmaxf(v, __shfl_xor(v, 1, 32));
  v = fmaxf(v, __shfl_xor(v, 2, 32));
  v = fmaxf(v, __shfl_xor(v, 4, 32));
  v = fmaxf(v, __shfl_xor(v, 8, 32));
  return v;
}
__device__ __forceinline__ float red_sum16(float v) {
  v += __shfl_xor(v, 1, 32);
  v += __shfl_xor(v, 2, 32);
  v += __shfl_xor(v, 4, 32);
  v += __shfl_xor(v, 8, 32);
  return v;
}

// ------------------------------- GEMM kernel --------------------------------
// C[M=4096, N=1024] = A[M,K=1024] @ W[K,N] + bias, bf16 WMMA / f32 acc.
// Block tile 128x128, 8 waves, each wave 32x64 (2x4 of 16x16) -> 8 WMMA/step.
// A_F32:    A read as f32 and converted to bf16 (QKV projections),
//           else A read as bf16 (output projection).
// HEAD_OUT: write bf16 into [B,H,S,dh] head-split layout with `scale`,
//           else write f32 row-major to d_out.
template <bool A_F32, bool HEAD_OUT>
__global__ __launch_bounds__(256) void gemm_kernel(const void* __restrict__ Aptr,
                                                   const float* __restrict__ W,
                                                   const float* __restrict__ bias,
                                                   void* __restrict__ Out,
                                                   float scale) {
  __shared__ __align__(16) __bf16 Asm[128][40];    // [m][k]
  __shared__ __align__(16) __bf16 BTsm[128][40];   // [n][k] (transposed)

  const int t    = threadIdx.x;
  const int wv   = t >> 5;
  const int lane = t & 31;
  const int half = lane >> 4, r = lane & 15;
  const int wm = wv & 3;          // 4 waves along M (32 rows each)
  const int wn = wv >> 2;         // 2 waves along N (64 cols each)
  const int m0 = blockIdx.y * 128;
  const int n0 = blockIdx.x * 128;

  v8f acc[2][4];
#pragma unroll
  for (int i = 0; i < 2; ++i)
#pragma unroll
    for (int j = 0; j < 4; ++j) acc[i][j] = (v8f){};

  for (int k0 = 0; k0 < DMODEL; k0 += 32) {
    // ---- stage A tile 128x32 (convert f32 -> bf16 if needed) ----
#pragma unroll
    for (int i = 0; i < 16; ++i) {
      const int idx = t + i * 256;              // 0..4095
      const int row = idx >> 5, col = idx & 31;
      float v;
      if constexpr (A_F32)
        v = ((const float*)Aptr)[(size_t)(m0 + row) * DMODEL + k0 + col];
      else
        v = (float)((const __bf16*)Aptr)[(size_t)(m0 + row) * DMODEL + k0 + col];
      Asm[row][col] = (__bf16)v;
    }
    // ---- stage W tile 32x128, transposed into BTsm[n][k] ----
#pragma unroll
    for (int i = 0; i < 16; ++i) {
      const int idx = t + i * 256;              // 0..4095
      const int kk = idx >> 7, nn = idx & 127;
      BTsm[nn][kk] = (__bf16)W[(size_t)(k0 + kk) * DMODEL + n0 + nn];
    }
    __syncthreads();

    if (t == 0 && k0 + 32 < DMODEL)             // gfx1250 global_prefetch path
      __builtin_prefetch(&W[(size_t)(k0 + 32) * DMODEL + n0], 0, 1);

    v16bf af[2], bfr[4];
#pragma unroll
    for (int i = 0; i < 2; ++i)
      af[i] = load_frag_a(&Asm[0][0], 40, wm * 32 + i * 16, 0);
#pragma unroll
    for (int j = 0; j < 4; ++j)
      bfr[j] = load_frag_b(&BTsm[0][0], 40, wn * 64 + j * 16, 0);
#pragma unroll
    for (int i = 0; i < 2; ++i)
#pragma unroll
      for (int j = 0; j < 4; ++j)
        acc[i][j] = wmma_bf16(af[i], bfr[j], acc[i][j]);
    __syncthreads();
  }

  // ---- epilogue: C row = half*8 + jj (VGPR jj), col = lane&15 ----
#pragma unroll
  for (int i = 0; i < 2; ++i) {
#pragma unroll
    for (int j = 0; j < 4; ++j) {
      const int col = n0 + wn * 64 + j * 16 + r;
      const float bcol = bias[col];
#pragma unroll
      for (int jj = 0; jj < 8; ++jj) {
        const int row = m0 + wm * 32 + i * 16 + half * 8 + jj;
        float v = acc[i][j][jj] + bcol;
        if constexpr (HEAD_OUT) {
          v *= scale;
          const int b = row >> 11, s = row & 2047;
          const int h = col >> 6, d = col & 63;
          ((__bf16*)Out)[(((size_t)(b * NHEAD + h)) * SEQ + s) * HDIM + d] =
              (__bf16)v;
        } else {
          ((float*)Out)[(size_t)row * DMODEL + col] = v;
        }
      }
    }
  }
}

// ---------------------------- Flash attention -------------------------------
// One block = 64 q-rows of one (b,h); 4 waves, each wave owns 16 q-rows.
// Qh/Kh/Vh: bf16 [B*H][S][64]; softmax scale already folded into Qh.
// Obuf: bf16 [B][S][H*64] for the output projection.
// Q and K tiles are staged with async Global->LDS B128 copies (ASYNCcnt);
// V is staged transposed (scattered LDS writes), so it uses the sync path.
#define LDT 80   // LDS row stride (elems): 160 B rows, 16 B aligned chunks

__global__ __launch_bounds__(128) void attn_kernel(const __bf16* __restrict__ Qh,
                                                   const __bf16* __restrict__ Kh,
                                                   const __bf16* __restrict__ Vh,
                                                   __bf16* __restrict__ Obuf) {
  __shared__ __align__(16) __bf16 Qsm[64][LDT];   // [q][dh]
  __shared__ __align__(16) __bf16 Ksm[64][LDT];   // [key][dh] (== Bt for Q@K^T)
  __shared__ __align__(16) __bf16 VTsm[64][LDT];  // [dh][key] (== Bt for P@V)
  __shared__ __align__(16) __bf16 Psm[64][LDT];   // [q][key], per-wave slab

  const int qt = blockIdx.x;      // 0..31 (q tile)
  const int bh = blockIdx.y;      // 0..31 (b*16 + h)
  const int t  = threadIdx.x;
  const int wv = t >> 5;
  const int lane = t & 31;
  const int half = lane >> 4, r = lane & 15;
  const size_t base_bh = (size_t)bh * SEQ * HDIM;

  // ---- stage Q tile once (async, 512 x 16B chunks / 128 threads) ----
#pragma unroll
  for (int i = 0; i < 4; ++i) {
    const int c = t + i * 128;                  // chunk id 0..511
    const int row = c >> 3, cc = (c & 7) * 8;   // 8 chunks of 8 bf16 per row
    copy16_to_lds(&Qsm[row][cc],
                  Qh + base_bh + (size_t)(qt * 64 + row) * HDIM + cc);
  }
  wait_async_copies();
  __syncthreads();

  v16bf aq[2];
  aq[0] = load_frag_a(&Qsm[0][0], LDT, wv * 16, 0);
  aq[1] = load_frag_a(&Qsm[0][0], LDT, wv * 16, 32);

  float mrow[8], lrow[8];
  v8f oacc[4];
#pragma unroll
  for (int j = 0; j < 8; ++j) { mrow[j] = -__builtin_inff(); lrow[j] = 0.f; }
#pragma unroll
  for (int ct = 0; ct < 4; ++ct) oacc[ct] = (v8f){};

  for (int kb = 0; kb < SEQ / 64; ++kb) {
    __syncthreads();   // previous iteration's K/V reads done before restage
    const size_t kvoff = base_bh + (size_t)(kb * 64) * HDIM;
    // K: async contiguous copy into [key][dh]
#pragma unroll
    for (int i = 0; i < 4; ++i) {
      const int c = t + i * 128;
      const int row = c >> 3, cc = (c & 7) * 8;
      copy16_to_lds(&Ksm[row][cc], Kh + kvoff + (size_t)row * HDIM + cc);
    }
    // V: sync transpose into [dh][key]
#pragma unroll
    for (int i = 0; i < 32; ++i) {
      const int idx = t + i * 128;
      const int row = idx >> 6, col = idx & 63;  // row=key, col=dh
      VTsm[col][row] = Vh[kvoff + (size_t)row * HDIM + col];
    }
    wait_async_copies();
    __syncthreads();

    // ---- S = Q @ K^T : 4 key-column tiles x 2 k-steps ----
    v8f sf[4];
#pragma unroll
    for (int ct = 0; ct < 4; ++ct) {
      v8f s = (v8f){};
      s = wmma_bf16(aq[0], load_frag_b(&Ksm[0][0], LDT, ct * 16, 0),  s);
      s = wmma_bf16(aq[1], load_frag_b(&Ksm[0][0], LDT, ct * 16, 32), s);
      sf[ct] = s;
    }

    // ---- online softmax: each lane owns 8 rows (row = half*8 + j) ----
    float corr[8];
#pragma unroll
    for (int j = 0; j < 8; ++j) {
      float mx = fmaxf(fmaxf(sf[0][j], sf[1][j]), fmaxf(sf[2][j], sf[3][j]));
      mx = red_max16(mx);
      const float mnew = fmaxf(mrow[j], mx);
      corr[j] = __expf(mrow[j] - mnew);
      float sum = 0.f;
#pragma unroll
      for (int ct = 0; ct < 4; ++ct) {
        const float p = __expf(sf[ct][j] - mnew);
        sf[ct][j] = p;
        sum += p;
      }
      sum = red_sum16(sum);
      lrow[j] = lrow[j] * corr[j] + sum;
      mrow[j] = mnew;
    }

    // ---- rescale O accumulator, spill P to LDS in A-loadable layout ----
#pragma unroll
    for (int ct = 0; ct < 4; ++ct)
#pragma unroll
      for (int j = 0; j < 8; ++j) {
        oacc[ct][j] *= corr[j];
        Psm[wv * 16 + half * 8 + j][ct * 16 + r] = (__bf16)sf[ct][j];
      }
    __syncthreads();

    // ---- O += P @ V : contraction over 64 keys = 2 k-steps ----
#pragma unroll
    for (int ks = 0; ks < 2; ++ks) {
      const v16bf ap = load_frag_a(&Psm[0][0], LDT, wv * 16, ks * 32);
#pragma unroll
      for (int ct = 0; ct < 4; ++ct)
        oacc[ct] = wmma_bf16(ap, load_frag_b(&VTsm[0][0], LDT, ct * 16, ks * 32),
                             oacc[ct]);
    }
  }

  // ---- normalize by row sums, write [B][S][H*64] bf16 ----
  const int b = bh >> 4, h = bh & 15;
#pragma unroll
  for (int j = 0; j < 8; ++j) {
    const float rinv = (lrow[j] > 0.f) ? (1.f / lrow[j]) : 0.f;
    const int s = qt * 64 + wv * 16 + half * 8 + j;
    const size_t rowoff = ((size_t)b * SEQ + s) * DMODEL + (size_t)h * HDIM;
#pragma unroll
    for (int ct = 0; ct < 4; ++ct)
      Obuf[rowoff + ct * 16 + r] = (__bf16)(oacc[ct][j] * rinv);
  }
}

// ------------------------------- launcher -----------------------------------
extern "C" void kernel_launch(void* const* d_in, const int* in_sizes, int n_in,
                              void* d_out, int out_size, void* d_ws,
                              size_t ws_size, hipStream_t stream) {
  (void)in_sizes; (void)n_in; (void)out_size; (void)ws_size;
  const float* q  = (const float*)d_in[0];
  const float* k  = (const float*)d_in[1];
  const float* v  = (const float*)d_in[2];
  const float* Wq = (const float*)d_in[3];
  const float* bq = (const float*)d_in[4];
  const float* Wk = (const float*)d_in[5];
  const float* bk = (const float*)d_in[6];
  const float* Wv = (const float*)d_in[7];
  const float* bv = (const float*)d_in[8];
  const float* Wo = (const float*)d_in[9];
  const float* bo = (const float*)d_in[10];
  float* out = (float*)d_out;

  const size_t HBUF = (size_t)BATCH * NHEAD * SEQ * HDIM;   // 4,194,304 elems
  __bf16* Qh = (__bf16*)d_ws;
  __bf16* Kh = Qh + HBUF;
  __bf16* Vh = Kh + HBUF;
  __bf16* Ob = Vh + HBUF;                                   // [B][S][D]

  const dim3 gg(DMODEL / 128, MROWS / 128);                 // (8, 32)
  const float qscale = 0.125f;                              // 1/sqrt(64)

  gemm_kernel<true, true><<<gg, 256, 0, stream>>>(q, Wq, bq, Qh, qscale);
  gemm_kernel<true, true><<<gg, 256, 0, stream>>>(k, Wk, bk, Kh, 1.0f);
  gemm_kernel<true, true><<<gg, 256, 0, stream>>>(v, Wv, bv, Vh, 1.0f);

  attn_kernel<<<dim3(SEQ / 64, BATCH * NHEAD), 128, 0, stream>>>(Qh, Kh, Vh, Ob);

  gemm_kernel<false, false><<<gg, 256, 0, stream>>>(Ob, Wo, bo, out, 1.0f);
}
